// FusedLoRAQKV_17368847745269
// MI455X (gfx1250) — compile-verified, matrix operand order checked by reading, not run on hardware
//
#include <hip/hip_runtime.h>
#include <hip/hip_bf16.h>

typedef __attribute__((ext_vector_type(16))) __bf16 v16bf;
typedef __attribute__((ext_vector_type(4)))  __bf16 v4bf;
typedef __attribute__((ext_vector_type(8)))  float  v8f;
typedef __attribute__((ext_vector_type(4)))  float  f32x4;

#define H_DIM   4096
#define M_TOT   8192
#define N_DIM   4096
#define R_DIM   16
#define LORA_SCALE 2.0f
#define BM 128
#define BN 128
#define BK 32
#define KTILES (H_DIM / BK)

// ---------------------------------------------------------------------------
// Kernel 1: XAs[p][m][r] = SCALING * sum_h X[m][h] * A_p[r][h]
// ---------------------------------------------------------------------------
__global__ __launch_bounds__(256) void lora_xa_kernel(
    const float* __restrict__ X,
    const float* __restrict__ A0, const float* __restrict__ A1,
    const float* __restrict__ A2,
    float* __restrict__ XAs)
{
    const int m   = blockIdx.x;
    const int p   = blockIdx.y;
    const int tid = threadIdx.x;
    const float* __restrict__ A = (p == 0) ? A0 : ((p == 1) ? A1 : A2);

    __shared__ float red[R_DIM][256];

    float acc[R_DIM];
    #pragma unroll
    for (int r = 0; r < R_DIM; ++r) acc[r] = 0.0f;

    const float* xrow = X + (size_t)m * H_DIM;
    for (int h = tid; h < H_DIM; h += 256) {
        float x = xrow[h];
        #pragma unroll
        for (int r = 0; r < R_DIM; ++r)
            acc[r] = fmaf(x, A[(size_t)r * H_DIM + h], acc[r]);
    }
    #pragma unroll
    for (int r = 0; r < R_DIM; ++r) red[r][tid] = acc[r];
    __syncthreads();

    for (int s = 128; s > 0; s >>= 1) {
        if (tid < s) {
            #pragma unroll
            for (int r = 0; r < R_DIM; ++r) red[r][tid] += red[r][tid + s];
        }
        __syncthreads();
    }
    if (tid < R_DIM)
        XAs[((size_t)p * M_TOT + m) * R_DIM + tid] = LORA_SCALE * red[tid][0];
}

// ---------------------------------------------------------------------------
// Kernel 2: fused Y_p = X*W_p^T + XAs_p * B_p^T  via split-bf16 WMMA.
// Block: 256 threads (8 waves). Block tile 128x128, K-tile 32.
// Wave grid 2(M) x 4(N); wave tile 64x32 -> 4x2 accum tiles of 16x16.
// Double-buffered LDS (one barrier / K-iter) + register prefetch of the
// next K-tile so global_load latency overlaps the 24 WMMAs.
// ---------------------------------------------------------------------------
__global__ __launch_bounds__(256) void lora_qkv_gemm_kernel(
    const float* __restrict__ X,
    const float* __restrict__ W0, const float* __restrict__ W1,
    const float* __restrict__ W2,
    const float* __restrict__ B0, const float* __restrict__ B1,
    const float* __restrict__ B2,
    const float* __restrict__ XAs,
    float* __restrict__ Y)
{
    // Fragment-swizzled tiles: [buffer][8 subtiles * 512 halfs]
    __shared__ __align__(32) __bf16 sAhi[2][8 * 512];
    __shared__ __align__(32) __bf16 sAlo[2][8 * 512];
    __shared__ __align__(32) __bf16 sBhi[2][8 * 512];
    __shared__ __align__(32) __bf16 sBlo[2][8 * 512];
    __shared__ __align__(16) float  sXA[BM * R_DIM];   // scaled X*A^T tile
    __shared__ __align__(16) float  sBl[BN * R_DIM];   // lora B tile

    const int tid   = threadIdx.x;
    const int p     = blockIdx.z;
    const int mBase = blockIdx.y * BM;
    const int nBase = blockIdx.x * BN;

    const float* __restrict__ W  = (p == 0) ? W0 : ((p == 1) ? W1 : W2);
    const float* __restrict__ Bl = (p == 0) ? B0 : ((p == 1) ? B1 : B2);

    // ---- stage LoRA epilogue tiles (128x16 each) into LDS -----------------
    #pragma unroll
    for (int i = 0; i < 8; i += 4) {
        int idx = tid * 8 + i;                  // 0..2047, 16B aligned
        int row = idx >> 4;
        int r   = idx & 15;
        *(f32x4*)&sXA[idx] =
            *(const f32x4*)&XAs[((size_t)p * M_TOT + mBase + row) * R_DIM + r];
        *(f32x4*)&sBl[idx] =
            *(const f32x4*)&Bl[(size_t)(nBase + row) * R_DIM + r];
    }

    const int lane  = tid & 31;
    const int wave  = tid >> 5;
    const int waveM = wave >> 2;   // 0..1
    const int waveN = wave & 3;    // 0..3

    v8f acc[4][2] = {};

    // tile-load assignment: 256 threads cover 128 rows x 2 k-halves
    const int rr   = tid & 127;              // row within tile
    const int kq   = (tid >> 7) << 4;        // 0 or 16
    const int ssub = rr >> 4;                // subtile 0..7
    const int mo   = rr & 15;
    const float* xsrc = X + (size_t)(mBase + rr) * H_DIM + kq;
    const float* wsrc = W + (size_t)(nBase + rr) * H_DIM + kq;

    // precomputed swizzled LDS half-offsets for the 4 quads this thread stores
    int stoff[4];
    #pragma unroll
    for (int i = 0; i < 4; ++i) {
        int kl   = kq + 4 * i;                       // 0..31, %4==0
        int lane_s = mo + (((kl >> 3) & 1) << 4);
        int hh     = (kl & 7) + ((kl >> 4) << 3);    // 4 consecutive halfs
        stoff[i]   = ssub * 512 + lane_s * 16 + hh;  // 8B aligned
    }

    // ---- prefetch K-tile 0 into registers ---------------------------------
    f32x4 xr[4], wr[4];
    #pragma unroll
    for (int i = 0; i < 4; ++i) {
        xr[i] = *(const f32x4*)(xsrc + 4 * i);
        wr[i] = *(const f32x4*)(wsrc + 4 * i);
    }

    for (int kt = 0; kt < KTILES; ++kt) {
        const int buf = kt & 1;

        // ---- convert prefetched regs -> split bf16 -> LDS[buf] -----------
        #pragma unroll
        for (int i = 0; i < 4; ++i) {
            v4bf xhi, xlo, whi, wlo;
            #pragma unroll
            for (int e = 0; e < 4; ++e) {
                __bf16 h1 = (__bf16)xr[i][e];
                xhi[e] = h1;
                xlo[e] = (__bf16)(xr[i][e] - (float)h1);
                __bf16 h2 = (__bf16)wr[i][e];
                whi[e] = h2;
                wlo[e] = (__bf16)(wr[i][e] - (float)h2);
            }
            *(v4bf*)&sAhi[buf][stoff[i]] = xhi;
            *(v4bf*)&sAlo[buf][stoff[i]] = xlo;
            *(v4bf*)&sBhi[buf][stoff[i]] = whi;
            *(v4bf*)&sBlo[buf][stoff[i]] = wlo;
        }
        __syncthreads();

        // ---- issue global loads for the NEXT K-tile (overlaps WMMA) ------
        if (kt + 1 < KTILES) {
            const float* xn = xsrc + (size_t)(kt + 1) * BK;
            const float* wn = wsrc + (size_t)(kt + 1) * BK;
            #pragma unroll
            for (int i = 0; i < 4; ++i) {
                xr[i] = *(const f32x4*)(xn + 4 * i);
                wr[i] = *(const f32x4*)(wn + 4 * i);
            }
        }

        // ---- fragment fetch: one 32B contiguous run per lane -------------
        v16bf ahi[4], alo[4], bhi[2], blo[2];
        #pragma unroll
        for (int mt = 0; mt < 4; ++mt) {
            int off = (waveM * 4 + mt) * 512 + lane * 16;
            ahi[mt] = *(const v16bf*)&sAhi[buf][off];
            alo[mt] = *(const v16bf*)&sAlo[buf][off];
        }
        #pragma unroll
        for (int nt = 0; nt < 2; ++nt) {
            int off = (waveN * 2 + nt) * 512 + lane * 16;
            bhi[nt] = *(const v16bf*)&sBhi[buf][off];
            blo[nt] = *(const v16bf*)&sBlo[buf][off];
        }

        // ---- 3-term split-bf16 WMMA: hi*hi + hi*lo + lo*hi ---------------
        #pragma unroll
        for (int mt = 0; mt < 4; ++mt) {
            #pragma unroll
            for (int nt = 0; nt < 2; ++nt) {
                v8f c = acc[mt][nt];
                c = __builtin_amdgcn_wmma_f32_16x16x32_bf16(
                        false, ahi[mt], false, bhi[nt], (short)0, c, false, false);
                c = __builtin_amdgcn_wmma_f32_16x16x32_bf16(
                        false, ahi[mt], false, blo[nt], (short)0, c, false, false);
                c = __builtin_amdgcn_wmma_f32_16x16x32_bf16(
                        false, alo[mt], false, bhi[nt], (short)0, c, false, false);
                acc[mt][nt] = c;
            }
        }
    }

    // ---- epilogue: add rank-16 LoRA term, store ---------------------------
    // C layout: VGPR j, lane l -> M = j + 8*(l>>4), N = l&15 (per 16x16 tile)
    const int mwb = waveM * 64;
    const int nwb = waveN * 32;
    #pragma unroll
    for (int mt = 0; mt < 4; ++mt) {
        #pragma unroll
        for (int j = 0; j < 8; ++j) {
            int ml = mwb + mt * 16 + ((lane >> 4) << 3) + j;
            const float* xa = &sXA[ml * R_DIM];
            #pragma unroll
            for (int nt = 0; nt < 2; ++nt) {
                int nl = nwb + nt * 16 + (lane & 15);
                const float* br = &sBl[nl * R_DIM];
                float sum = acc[mt][nt][j];
                #pragma unroll
                for (int r = 0; r < R_DIM; ++r)
                    sum = fmaf(xa[r], br[r], sum);
                Y[((size_t)p * M_TOT + (size_t)(mBase + ml)) * N_DIM
                  + (size_t)(nBase + nl)] = sum;
            }
        }
    }
}

// ---------------------------------------------------------------------------
extern "C" void kernel_launch(void* const* d_in, const int* in_sizes, int n_in,
                              void* d_out, int out_size, void* d_ws, size_t ws_size,
                              hipStream_t stream)
{
    (void)in_sizes; (void)n_in; (void)out_size; (void)ws_size;

    const float* X  = (const float*)d_in[0];
    const float* QW = (const float*)d_in[1];
    const float* QA = (const float*)d_in[2];
    const float* QB = (const float*)d_in[3];
    const float* KW = (const float*)d_in[4];
    const float* KA = (const float*)d_in[5];
    const float* KB = (const float*)d_in[6];
    const float* VW = (const float*)d_in[7];
    const float* VA = (const float*)d_in[8];
    const float* VB = (const float*)d_in[9];

    float* XAs = (float*)d_ws;            // 3 * 8192 * 16 floats = 1.5 MB
    float* Y   = (float*)d_out;           // Q,K,V concatenated

    dim3 g1(M_TOT, 3, 1);
    lora_xa_kernel<<<g1, 256, 0, stream>>>(X, QA, KA, VA, XAs);

    dim3 g2(N_DIM / BN, M_TOT / BM, 3);   // (32, 64, 3)
    lora_qkv_gemm_kernel<<<g2, 256, 0, stream>>>(
        X, QW, KW, VW, QB, KB, VB, XAs, Y);
}
